// PatchEmbed_13735305413351
// MI455X (gfx1250) — matrix operand order repaired and checked by reading, not compiled, not run on hardware
//
#include <hip/hip_runtime.h>
#include <hip/hip_bf16.h>
#include <math.h>

// ---------------------------------------------------------------------------
// PatchEmbed (spherical unified-camera model) fused sampler + WMMA projection
// for gfx1250 (CDNA5, wave32).
//
// Pipeline:
//   prep_w_kernel : W [96x300] f32 -> Wh [96x320] f16 (zero-padded K) in d_ws
//   patch_embed   : per WG = (image b, quarter q of azimuth sectors)
//                   Phase A: bilinear-sample 16 patches x 100 coords, the 3
//                            channels share coordinates/weights -> LDS f16
//                   Phase B: v_wmma_f32_16x16x32_f16, K = 10 steps of 32;
//                            all 10 W fragments live in persistent VGPRs.
// Outputs: tokens [256,1024,96] f32 then D_s [256,32,10] f32, concatenated.
// ---------------------------------------------------------------------------

typedef __attribute__((ext_vector_type(16))) _Float16     v16h;
typedef __attribute__((ext_vector_type(8)))  float        v8f;
typedef __attribute__((ext_vector_type(4)))  unsigned int u32x4;

#define B_IMG   256
#define C_CH    3
#define HW_IMG  64
#define RC_     32
#define AC_     32
#define NR_     10
#define NA_     10
#define NS      (NR_ * NA_)    // 100 samples per patch
#define EMBED_  96
#define KDIM    300            // C*NR*NA
#define KPAD    320            // padded K (10 steps of 32)
#define NKS     (KPAD / 32)    // 10 K-steps
#define SA      328            // LDS A-tile stride in halves (rows 16B aligned)
#define P_TOT   (RC_ * AC_)    // 1024 patches
#define THETA_MAX 1.0471975511965976f   // pi/3

union FragH { u32x4 u[2]; v16h h; };

// --------------------------- W -> f16 prep ---------------------------------
__global__ __launch_bounds__(256)
void prep_w_kernel(const float* __restrict__ Wsrc, _Float16* __restrict__ Wh) {
    int idx = blockIdx.x * 256 + threadIdx.x;
    if (idx >= EMBED_ * KPAD) return;
    int e = idx / KPAD;
    int k = idx - e * KPAD;
    float v = (k < KDIM) ? Wsrc[e * KDIM + k] : 0.0f;
    Wh[idx] = (_Float16)v;
}

// --------------------------- fused main kernel -----------------------------
__global__ __launch_bounds__(192)
void patch_embed_kernel(const float* __restrict__ x,
                        const float* __restrict__ dist,
                        const _Float16* __restrict__ Wh,
                        const float* __restrict__ bias,
                        float* __restrict__ out) {
    __shared__ _Float16 featA[16 * SA];   // 16 patches x padded-K  (10.5 KB)
    __shared__ float    rL[RC_ * NR_];    // ring radii
    __shared__ float    cpL[8 * NA_];     // cos(phi) for this quarter's sectors
    __shared__ float    spL[8 * NA_];     // sin(phi)
    __shared__ float    biasL[EMBED_];

    const int tid = threadIdx.x;
    const int b   = blockIdx.x >> 2;      // image
    const int q   = blockIdx.x & 3;       // quarter of azimuth sectors

    const float xi = dist[b];                                   // dist is [1,B]
    const float f  = 32.0f * (xi + 0.5f) * 1.1547005383792515f; // R_MAX*(xi+cos60)/sin60

    // ring radii r(rc,nr) -> LDS; q==0 also writes the D_s output
    for (int i = tid; i < RC_ * NR_; i += 192) {
        float t  = ((float)i + 0.5f) * (THETA_MAX / (float)(RC_ * NR_));
        float rv = f * __sinf(t) / (xi + __cosf(t));
        rL[i] = rv;
        if (q == 0)
            out[(size_t)B_IMG * P_TOT * EMBED_ + (size_t)b * (RC_ * NR_) + i] = rv;
    }
    // azimuth trig for this quarter's 8 sectors
    for (int i = tid; i < 8 * NA_; i += 192) {
        int acl = i / NA_, na = i - acl * NA_;
        int ac  = q * 8 + acl;
        float p = ((float)(ac * NA_ + na) + 0.5f) *
                  (6.283185307179586f / (float)(AC_ * NA_));
        cpL[i] = __cosf(p);
        spL[i] = __sinf(p);
    }
    for (int i = tid; i < EMBED_; i += 192) biasL[i] = bias[i];
    // zero the K pad [300, SA) once; never rewritten afterwards
    for (int i = tid; i < 16 * (SA - KDIM); i += 192) {
        int m = i / (SA - KDIM), k = KDIM + (i - m * (SA - KDIM));
        featA[m * SA + k] = (_Float16)0.0f;
    }
    __syncthreads();

    const int lane = tid & 31;
    const int wv   = tid >> 5;            // 0..5 -> N-tile of EMBED
    const int hh   = lane >> 4;           // lane half-group
    const int lrow = lane & 15;           // M row (A/C) or N col (B/C)
    const float* imgB = x + (size_t)b * C_CH * HW_IMG * HW_IMG;
    const _Float16* wrow = Wh + (size_t)(wv * 16 + lrow) * KPAD;
    const _Float16* arow = &featA[lrow * SA];

    for (int mt = 0; mt < 16; ++mt) {
        const int pbase = q * 256 + mt * 16;

        // ---- Phase A: 16 patches x 100 coords; 3 channels share coords ------
        for (int flat = tid; flat < 16 * NS; flat += 192) {
            int m   = flat / NS;                   // patch row in tile
            int s   = flat - m * NS;               // sample index (nr*10+na)
            int nr  = s / NA_, na = s - nr * NA_;
            int p   = pbase + m;
            int rc  = p & 31;
            int acl = (p >> 5) - q * 8;

            float rv = rL[rc * NR_ + nr];
            float cp = cpL[acl * NA_ + na];
            float sp = spL[acl * NA_ + na];
            // grid_sample align_corners=True: gx = (sx/32+1)*0.5*63
            float gx = rv * cp * 0.984375f + 31.5f;
            float gy = -rv * sp * 0.984375f + 31.5f;

            float x0f = floorf(gx), y0f = floorf(gy);
            float wx1 = gx - x0f, wy1 = gy - y0f;
            float wx0 = 1.0f - wx1, wy0 = 1.0f - wy1;
            int ix0 = (int)x0f, iy0 = (int)y0f;
            int ix1 = ix0 + 1,  iy1 = iy0 + 1;
            bool vx0 = (ix0 >= 0) & (ix0 < HW_IMG);
            bool vx1 = (ix1 >= 0) & (ix1 < HW_IMG);
            bool vy0 = (iy0 >= 0) & (iy0 < HW_IMG);
            bool vy1 = (iy1 >= 0) & (iy1 < HW_IMG);
            bool m00 = vx0 & vy0, m10 = vx1 & vy0;
            bool m01 = vx0 & vy1, m11 = vx1 & vy1;

            float w00 = wx0 * wy0, w10 = wx1 * wy0;
            float w01 = wx0 * wy1, w11 = wx1 * wy1;
            int i00 = iy0 * HW_IMG + ix0;          // tap offsets, channel 0
            int i10 = iy0 * HW_IMG + ix1;
            int i01 = iy1 * HW_IMG + ix0;
            int i11 = iy1 * HW_IMG + ix1;

            _Float16* frow = &featA[m * SA + s];
            #pragma unroll
            for (int c = 0; c < C_CH; ++c) {
                const float* img = imgB + c * (HW_IMG * HW_IMG);
                float v00 = m00 ? img[i00] : 0.0f;
                float v10 = m10 ? img[i10] : 0.0f;
                float v01 = m01 ? img[i01] : 0.0f;
                float v11 = m11 ? img[i11] : 0.0f;
                float acc = v00 * w00 + v10 * w10 + v01 * w01 + v11 * w11;
                frow[c * NS] = (_Float16)acc;      // feat[m][c*100 + s]
            }
        }
        __syncthreads();

        // ---- Phase B: 16x16x32 f16 WMMA over K = 10 steps -------------------
        // (EXEC all-ones: whole 192-thread block executes this uniformly)
        v8f cacc = {};
        #pragma unroll
        for (int ks = 0; ks < NKS; ++ks) {
            FragH a, bf;
            // A 16x32 f16 layout: per lane two contiguous 8-half runs
            int abase = ks * 32 + hh * 8;
            a.u[0] = *(const u32x4*)(arow + abase);        // ds_load_b128
            a.u[1] = *(const u32x4*)(arow + abase + 16);   // ds_load_b128
            // B 32x16 f16 fragment: loop-invariant w.r.t. mt -> compiler
            // hoists all 10 fragments into persistent VGPRs.
            int bbase = ks * 32 + hh * 8;
            bf.u[0] = *(const u32x4*)(wrow + bbase);       // global_load_b128
            bf.u[1] = *(const u32x4*)(wrow + bbase + 16);  // global_load_b128
            cacc = __builtin_amdgcn_wmma_f32_16x16x32_f16(
                       false, a.h, false, bf.h, (short)0, cacc, false, false);
        }
        __syncthreads();   // protect featA before next Phase A overwrite

        // ---- epilogue: bias + transposed token store ------------------------
        float bv = biasL[wv * 16 + lrow];
        int mb = hh * 8;
        #pragma unroll
        for (int i = 0; i < 8; ++i) {
            int p   = pbase + mb + i;
            int rc  = p & 31;
            int acg = p >> 5;
            int t   = rc * AC_ + acg;                   // token transpose
            out[((size_t)b * P_TOT + t) * EMBED_ + wv * 16 + lrow] = cacc[i] + bv;
        }
    }
}

// ---------------------------------------------------------------------------
extern "C" void kernel_launch(void* const* d_in, const int* in_sizes, int n_in,
                              void* d_out, int out_size, void* d_ws, size_t ws_size,
                              hipStream_t stream) {
    const float* x    = (const float*)d_in[0];   // [256,3,64,64]
    const float* dist = (const float*)d_in[1];   // [1,256]
    const float* W    = (const float*)d_in[2];   // [96,300]
    const float* bias = (const float*)d_in[3];   // [96]
    float*       out  = (float*)d_out;           // tokens ++ D_s
    _Float16*    Wh   = (_Float16*)d_ws;         // 96*320 f16 = 60 KB scratch

    // W -> padded f16
    {
        int total  = EMBED_ * KPAD;
        int blocks = (total + 255) / 256;
        prep_w_kernel<<<blocks, 256, 0, stream>>>(W, Wh);
    }
    // fused sample + WMMA projection: 1024 WGs x 192 threads (6 waves)
    patch_embed_kernel<<<B_IMG * 4, 192, 0, stream>>>(x, dist, Wh, bias, out);
}
